// CrystalGraphEncoder_7275674599727
// MI455X (gfx1250) — matrix operand order repaired and checked by reading, not compiled, not run on hardware
//
#include <hip/hip_runtime.h>
#include <hip/hip_bf16.h>

// ---------------------------------------------------------------------------
// CDNA5 (gfx1250) crystal-graph encoder.
// Edge & node GEMMs on v_wmma_f32_16x16x32_bf16 (f32 accumulate).
// x kept in f32 (precision path) + bf16 mirror (WMMA A-operand gathers).
// Weights pre-packed to bf16 WMMA B-layout, staged in LDS per workgroup.
// ---------------------------------------------------------------------------

typedef __attribute__((ext_vector_type(16))) __bf16 v16bf;
typedef __attribute__((ext_vector_type(8)))  float  v8f;

union ATile {
    v16bf v;
    __bf16 b[16];
    uint4 q[2];
};

// ---------------------------------------------------------------------------
// Zero-fill helper (graph-capture safe scratch init)
// ---------------------------------------------------------------------------
__global__ void zero_kernel(float* __restrict__ p, int n) {
    int t = blockIdx.x * blockDim.x + threadIdx.x;
    if (t < n) p[t] = 0.0f;
}

// ---------------------------------------------------------------------------
// Embedding gather: x[a][h] = emb_table[z[a]][h]  (f32 + bf16 mirror)
// ---------------------------------------------------------------------------
__global__ void embed_kernel(const int* __restrict__ z,
                             const float* __restrict__ emb,
                             float* __restrict__ x,
                             __bf16* __restrict__ xb,
                             int n64, int nElems) {
    int t = blockIdx.x * blockDim.x + threadIdx.x;
    if (t >= n64) return;
    int a = t >> 6;
    int h = t & 63;
    int zi = z[a];
    if (zi < 0) zi = 0;
    if (zi >= nElems) zi = nElems - 1;
    float v = emb[zi * 64 + h];
    x[t] = v;
    xb[t] = (__bf16)v;
}

// ---------------------------------------------------------------------------
// Repack W_gate [3][128][64] and W_lin [3][64][64] (f32 row-major) into bf16
// WMMA-B operand order:
//   block(kt,nt): 32 lanes x 16 bf16.  Lane l holds column N = nt*16 + (l&15),
//   K = kt*32 + (jj<8 ? 0 : 16) + (l>=16 ? 8 : 0) + (jj&7)  (jj = 0..15 == 2*vgpr + half).
// Block id = kt*4 + nt.  pWg: kt 0..3, pWl: kt 0..1.
// ---------------------------------------------------------------------------
__global__ void pack_weights(const float* __restrict__ W_gate,
                             const float* __restrict__ W_lin,
                             __bf16* __restrict__ pWg,
                             __bf16* __restrict__ pWl) {
    int t = blockIdx.x * blockDim.x + threadIdx.x;
    const int wgTotal = 3 * 16 * 512;
    const int wlTotal = 3 * 8 * 512;
    if (t < wgTotal) {
        int layer = t / (16 * 512);
        int rem   = t % (16 * 512);
        int block = rem / 512;        // kt*4 + nt
        int idx   = rem % 512;        // lane*16 + jj
        int lane  = idx >> 4;
        int jj    = idx & 15;
        int kt = block >> 2, nt = block & 3;
        int K = kt * 32 + ((jj >= 8) ? 16 : 0) + ((lane >= 16) ? 8 : 0) + (jj & 7);
        int N = nt * 16 + (lane & 15);
        pWg[t] = (__bf16)W_gate[layer * 128 * 64 + K * 64 + N];
    } else if (t < wgTotal + wlTotal) {
        int u = t - wgTotal;
        int layer = u / (8 * 512);
        int rem   = u % (8 * 512);
        int block = rem / 512;        // kt*4 + nt (kt in 0..1)
        int idx   = rem % 512;
        int lane  = idx >> 4;
        int jj    = idx & 15;
        int kt = block >> 2, nt = block & 3;
        int K = kt * 32 + ((jj >= 8) ? 16 : 0) + ((lane >= 16) ? 8 : 0) + (jj & 7);
        int N = nt * 16 + (lane & 15);
        pWl[u] = (__bf16)W_lin[layer * 64 * 64 + K * 64 + N];
    }
}

// Load one B tile (32 bytes/lane) from an LDS-resident packed block.
__device__ __forceinline__ void load_b_lds(const __bf16* __restrict__ base,
                                           int lane, ATile& B) {
    const uint4* bp = (const uint4*)base + lane * 2;
    B.q[0] = bp[0];
    B.q[1] = bp[1];
}

// Load one A tile (16 rows x 32 K slice for this lane) from the bf16 x mirror.
// 16-bit A layout: lane<16 -> K {0..7,16..23}; lane>=16 -> K {8..15,24..31}.
__device__ __forceinline__ void load_a_bf16(const __bf16* __restrict__ src, ATile& A) {
    const uint4* p = (const uint4*)src;    // src is 16B aligned
    A.q[0] = p[0];        // K offsets +0..7
    A.q[1] = p[2];        // K offsets +16..23 (i.e. +32 bytes)
}

// ---------------------------------------------------------------------------
// Edge kernel: one wave == 16 edges.
//   gate = sigmoid([x[row] | x[col]] @ Wg + bg)   (K=128 -> 4 k-tiles)
//   msg  = gate * (x[col] @ Wl + bl)              (K=64  -> 2 k-tiles, reuse A)
//   atomicAdd(agg[row], msg)
// ---------------------------------------------------------------------------
__global__ void edge_kernel(const __bf16* __restrict__ xb,
                            const int* __restrict__ row,
                            const int* __restrict__ col,
                            const __bf16* __restrict__ pWg,  // 16 blocks x 512
                            const __bf16* __restrict__ pWl,  // 8 blocks x 512
                            const float* __restrict__ b_gate, // [64]
                            const float* __restrict__ b_lin,  // [64]
                            float* __restrict__ agg,
                            int nEdges, int nTiles) {
    __shared__ __align__(16) __bf16 sWg[16 * 512];   // 16 KB
    __shared__ __align__(16) __bf16 sWl[8 * 512];    //  8 KB

    // Cooperative stage of this layer's packed weights into LDS.
    {
        uint4* dg = (uint4*)sWg;  const uint4* gg = (const uint4*)pWg;
        uint4* dl = (uint4*)sWl;  const uint4* gl = (const uint4*)pWl;
        for (int i = threadIdx.x; i < 1024; i += 256) dg[i] = gg[i];   // 16 KB
        for (int i = threadIdx.x; i < 512;  i += 256) dl[i] = gl[i];   //  8 KB
    }
    __syncthreads();

    int wave = (int)((blockIdx.x * blockDim.x + threadIdx.x) >> 5);
    int lane = threadIdx.x & 31;
    if (wave >= nTiles) return;          // wave-uniform: EXEC stays all-1s below

    int m  = lane & 15;
    int hi = lane >> 4;                  // 0 or 1
    int e = wave * 16 + m;
    if (e >= nEdges) e = nEdges - 1;     // clamp (stores are guarded later)
    int r = row[e];
    int c = col[e];

    // A tiles: kt 0..1 from x[row] (gate K 0..63), kt 2..3 from x[col]
    // (gate K 64..127, doubling as the message-GEMM A operand).
    ATile A[4];
#pragma unroll
    for (int kt = 0; kt < 4; ++kt) {
        int node = (kt < 2) ? r : c;
        load_a_bf16(xb + node * 64 + (kt & 1) * 32 + hi * 8, A[kt]);
    }

    v8f g[4], mm[4];
#pragma unroll
    for (int nt = 0; nt < 4; ++nt) {
        v8f ga = {};
        v8f ma = {};
#pragma unroll
        for (int kt = 0; kt < 4; ++kt) {
            ATile B;
            load_b_lds(sWg + (kt * 4 + nt) * 512, lane, B);
            ga = __builtin_amdgcn_wmma_f32_16x16x32_bf16(
                     false, A[kt].v, false, B.v, (short)0, ga, false, false);
        }
#pragma unroll
        for (int kt = 0; kt < 2; ++kt) {
            ATile B;
            load_b_lds(sWl + (kt * 4 + nt) * 512, lane, B);
            ma = __builtin_amdgcn_wmma_f32_16x16x32_bf16(
                     false, A[2 + kt].v, false, B.v, (short)0, ma, false, false);
        }
        g[nt]  = ga;
        mm[nt] = ma;
    }

    // D layout: element (vgpr rv, lane) -> edge row rv + hi*8, col nt*16 + (lane&15).
    // Lane (rv + hi*8) already holds row[wave*16 + rv + hi*8] in r: shuffle it over.
    int rows8[8];
    bool ok8[8];
#pragma unroll
    for (int rv = 0; rv < 8; ++rv) {
        int e2 = wave * 16 + rv + hi * 8;
        ok8[rv] = (e2 < nEdges);
        rows8[rv] = __shfl(r, hi * 8 + rv, 32);
    }
    int nlo = m;
#pragma unroll
    for (int nt = 0; nt < 4; ++nt) {
        float bg = b_gate[nt * 16 + nlo];
        float bl = b_lin[nt * 16 + nlo];
#pragma unroll
        for (int rv = 0; rv < 8; ++rv) {
            if (ok8[rv]) {
                float gv = 1.0f / (1.0f + __expf(-(g[nt][rv] + bg)));
                float mv = gv * (mm[nt][rv] + bl);
                atomicAdd(agg + rows8[rv] * 64 + nt * 16 + nlo, mv);
            }
        }
    }
}

// ---------------------------------------------------------------------------
// Node update: x = relu(agg + x @ Wl + bl).  One wave == 16 atoms, in-place
// (each wave reads/writes only its own rows; A is consumed into registers
// before the stores).  Maintains the bf16 mirror for the next layer.
// ---------------------------------------------------------------------------
__global__ void node_update_kernel(float* __restrict__ x,
                                   __bf16* __restrict__ xb,
                                   const float* __restrict__ agg,
                                   const __bf16* __restrict__ pWl,
                                   const float* __restrict__ b_lin,
                                   int nAtoms, int nTiles) {
    __shared__ __align__(16) __bf16 sWl[8 * 512];    // 8 KB
    {
        uint4* dl = (uint4*)sWl;  const uint4* gl = (const uint4*)pWl;
        for (int i = threadIdx.x; i < 512; i += 256) dl[i] = gl[i];
    }
    __syncthreads();

    int wave = (int)((blockIdx.x * blockDim.x + threadIdx.x) >> 5);
    int lane = threadIdx.x & 31;
    if (wave >= nTiles) return;

    int m  = lane & 15;
    int hi = lane >> 4;
    int a = wave * 16 + m;
    if (a >= nAtoms) a = nAtoms - 1;

    ATile A[2];
#pragma unroll
    for (int kt = 0; kt < 2; ++kt)
        load_a_bf16(xb + a * 64 + kt * 32 + hi * 8, A[kt]);

    v8f acc[4];
#pragma unroll
    for (int nt = 0; nt < 4; ++nt) {
        v8f s = {};
#pragma unroll
        for (int kt = 0; kt < 2; ++kt) {
            ATile B;
            load_b_lds(sWl + (kt * 4 + nt) * 512, lane, B);
            s = __builtin_amdgcn_wmma_f32_16x16x32_bf16(
                    false, A[kt].v, false, B.v, (short)0, s, false, false);
        }
        acc[nt] = s;
    }

    int nlo = m;
#pragma unroll
    for (int nt = 0; nt < 4; ++nt) {
        float bl = b_lin[nt * 16 + nlo];
#pragma unroll
        for (int rv = 0; rv < 8; ++rv) {
            int a2 = wave * 16 + rv + hi * 8;
            if (a2 < nAtoms) {
                int off = a2 * 64 + nt * 16 + nlo;
                float v = agg[off] + acc[nt][rv] + bl;
                v = fmaxf(v, 0.0f);
                x[off]  = v;
                xb[off] = (__bf16)v;
            }
        }
    }
}

// ---------------------------------------------------------------------------
// Mean-pool partial sums over the f32 x (precision path)
// ---------------------------------------------------------------------------
__global__ void pool_reduce(const float* __restrict__ x,
                            float* __restrict__ pooled,
                            int nAtoms) {
    int t = blockIdx.x * blockDim.x + threadIdx.x;
    int n = t & 63;
    int stride = (gridDim.x * blockDim.x) >> 6;
    float s = 0.0f;
    for (int a = t >> 6; a < nAtoms; a += stride) s += x[a * 64 + n];
    atomicAdd(pooled + n, s);
}

// ---------------------------------------------------------------------------
// Head: h = relu(pooled/N @ W1 + b1); out = h @ W2 + b2.  Single block.
// ---------------------------------------------------------------------------
__global__ void head_kernel(const float* __restrict__ pooled,
                            const float* __restrict__ W1,
                            const float* __restrict__ b1,
                            const float* __restrict__ W2,
                            const float* __restrict__ b2,
                            float* __restrict__ out,
                            int nAtoms) {
    __shared__ float pm[64];
    __shared__ float h[64];
    int t = threadIdx.x;   // 128 threads
    if (t < 64) pm[t] = pooled[t] / (float)nAtoms;
    __syncthreads();
    if (t < 64) {
        float acc = b1[t];
        for (int k = 0; k < 64; ++k) acc += pm[k] * W1[k * 64 + t];
        h[t] = fmaxf(acc, 0.0f);
    }
    __syncthreads();
    if (t < 128) {
        float acc = b2[t];
        for (int k = 0; k < 64; ++k) acc += h[k] * W2[k * 128 + t];
        out[t] = acc;
    }
}

// ---------------------------------------------------------------------------
// Launcher
// ---------------------------------------------------------------------------
extern "C" void kernel_launch(void* const* d_in, const int* in_sizes, int n_in,
                              void* d_out, int out_size, void* d_ws, size_t ws_size,
                              hipStream_t stream) {
    const int*   z      = (const int*)d_in[0];
    const int*   eidx   = (const int*)d_in[3];
    const float* emb    = (const float*)d_in[4];
    const float* W_lin  = (const float*)d_in[5];
    const float* b_lin  = (const float*)d_in[6];
    const float* W_gate = (const float*)d_in[7];
    const float* b_gate = (const float*)d_in[8];
    const float* W1     = (const float*)d_in[9];
    const float* b1     = (const float*)d_in[10];
    const float* W2     = (const float*)d_in[11];
    const float* b2     = (const float*)d_in[12];
    float* out = (float*)d_out;

    int N = in_sizes[0];
    int E = in_sizes[3] / 2;
    int nElems = in_sizes[4] / 64;
    const int* row = eidx;
    const int* col = eidx + E;

    // Workspace carve-up (256B aligned regions)
    char* ws = (char*)d_ws;
    size_t off = 0;
    auto carve = [&](size_t bytes) -> void* {
        void* p = ws + off;
        off = (off + bytes + 255) & ~(size_t)255;
        return p;
    };
    float*  x      = (float*)carve((size_t)N * 64 * sizeof(float));
    float*  agg    = (float*)carve((size_t)N * 64 * sizeof(float));
    __bf16* xb     = (__bf16*)carve((size_t)N * 64 * sizeof(__bf16));
    float*  pooled = (float*)carve(64 * sizeof(float));
    __bf16* pWg    = (__bf16*)carve(3 * 16 * 512 * sizeof(__bf16));
    __bf16* pWl    = (__bf16*)carve(3 * 8 * 512 * sizeof(__bf16));
    (void)ws_size; (void)n_in; (void)out_size;

    const int TB = 256;

    // 1) embedding gather
    {
        int n64 = N * 64;
        embed_kernel<<<(n64 + TB - 1) / TB, TB, 0, stream>>>(z, emb, x, xb, n64, nElems);
    }

    // 2) weight repack (bf16, WMMA B-operand order)
    {
        int total = 3 * 16 * 512 + 3 * 8 * 512;
        pack_weights<<<(total + TB - 1) / TB, TB, 0, stream>>>(W_gate, W_lin, pWg, pWl);
    }

    // 3) message-passing layers
    int tilesE  = (E + 15) / 16;
    int tilesN  = (N + 15) / 16;
    int blocksE = (tilesE + 7) / 8;   // 8 waves (tiles) per 256-thread block
    int blocksN = (tilesN + 7) / 8;
    for (int l = 0; l < 3; ++l) {
        zero_kernel<<<((N * 64) + TB - 1) / TB, TB, 0, stream>>>(agg, N * 64);
        edge_kernel<<<blocksE, TB, 0, stream>>>(
            xb, row, col,
            pWg + (size_t)l * 16 * 512,
            pWl + (size_t)l * 8 * 512,
            b_gate + l * 64, b_lin + l * 64,
            agg, E, tilesE);
        node_update_kernel<<<blocksN, TB, 0, stream>>>(
            x, xb, agg,
            pWl + (size_t)l * 8 * 512,
            b_lin + l * 64,
            N, tilesN);
    }

    // 4) mean pool + head
    zero_kernel<<<1, 64, 0, stream>>>(pooled, 64);
    pool_reduce<<<512, TB, 0, stream>>>(x, pooled, N);
    head_kernel<<<1, 128, 0, stream>>>(pooled, W1, b1, W2, b2, out, N);
}
